// Polynormer_55757265437199
// MI455X (gfx1250) — matrix operand order repaired and checked by reading, not compiled, not run on hardware
//
// Polynormer forward for MI455X (gfx1250): bf16 WMMA GEMMs + fp32 accumulate,
// LDS-transposed split-K WMMA for kv = k^T v, wave32 row reductions for LN.
// Round 3: batch-preload ALL B tiles per K-step ahead of the WMMA burst so the
// compiler emits decreasing partial s_wait_dscnt (LDS reads overlap the XDL
// pipe) instead of a full wait before every WMMA.
#include <hip/hip_runtime.h>
#include <hip/hip_bf16.h>
#include <math.h>

#define DCH 128
#define GLAYERS 2
#define OUTC 40
#define BETA_C 0.9f
#define EPS_C 1e-5f

typedef __attribute__((ext_vector_type(16))) __bf16 v16bf;
typedef __attribute__((ext_vector_type(8)))  float  v8f;
typedef __attribute__((ext_vector_type(4)))  unsigned int u32x4;

union ABReg { u32x4 q[2]; v16bf v; };

__device__ __forceinline__ unsigned short f2bf(float f) {
  unsigned int u = __float_as_uint(f);
  u += 0x7FFFu + ((u >> 16) & 1u);          // round-to-nearest-even
  return (unsigned short)(u >> 16);
}
__device__ __forceinline__ float bf2f(unsigned short h) {
  return __uint_as_float(((unsigned int)h) << 16);
}
__device__ __forceinline__ float wred32(float x) {
#pragma unroll
  for (int off = 16; off > 0; off >>= 1) x += __shfl_xor(x, off, 32);
  return x;
}

// ---------------------------------------------------------------------------
// Generic fused GEMM: Y = act(Xbf16[N,128] @ W_fp32[128,out_cols] + bias)
// 8 waves/block, wave owns 16 rows x (NTILES*16) cols via v_wmma_f32_16x16x32_bf16.
// act: 0=none 1=relu 2=sigmoid. Yf (fp32, stride ldo) and/or Yb (bf16, stride 128).
// A fragments preloaded from global; all B tiles of a K-step preloaded from LDS
// before the WMMA burst (partial dscnt waits, DS/XDL overlap).
// ---------------------------------------------------------------------------
template <int NTILES>
__global__ __launch_bounds__(256) void k_gemm(
    const unsigned short* __restrict__ Xb, const float* __restrict__ W,
    const float* __restrict__ bias, float* __restrict__ Yf,
    unsigned short* __restrict__ Yb, int N, int out_cols, int ldo, int act) {
  constexpr int WSTR = 136;  // bf16 elems/row: 272B, 16B aligned, conflict-light
  __shared__ __align__(16) unsigned short wt[NTILES * 16 * WSTR];
  const int tid = threadIdx.x;
  const int wave = tid >> 5, lane = tid & 31;
  const int lhalf = lane >> 4;
  const int lrow  = lane & 15;
  const int r0 = blockIdx.x * 128 + wave * 16;

  // Preload ALL A fragments (wave's 16x128 bf16 block) before the barrier so
  // the MMA loop has no loadcnt dependence.
  const unsigned short* arow = Xb + (size_t)(r0 + lrow) * DCH;
  ABReg a[4];
#pragma unroll
  for (int ks = 0; ks < 4; ++ks) {
    a[ks].q[0] = *(const u32x4*)(arow + ks * 32 + lhalf * 8);
    a[ks].q[1] = *(const u32x4*)(arow + ks * 32 + 16 + lhalf * 8);
  }

  // Stage W transposed (wt[o][i]) as bf16; linear LDS writes, L2-resident reads.
  for (int idx = tid; idx < NTILES * 16 * DCH; idx += 256) {
    int o = idx >> 7, i = idx & 127;
    float v = (o < out_cols) ? W[(size_t)i * out_cols + o] : 0.f;
    wt[o * WSTR + i] = f2bf(v);
  }
  __syncthreads();

  v8f acc[NTILES];
#pragma unroll
  for (int t = 0; t < NTILES; ++t)
#pragma unroll
    for (int j = 0; j < 8; ++j) acc[t][j] = 0.f;

#pragma unroll
  for (int ks = 0; ks < 4; ++ks) {           // K = 128 = 4 x 32
    // Issue every B-tile load of this K-step before the first WMMA: DS returns
    // in order, so waits become partial (dscnt <= 2*(NTILES-1-t)).
    ABReg b[NTILES];
#pragma unroll
    for (int t = 0; t < NTILES; ++t) {
      const unsigned short* br =
          wt + (t * 16 + lrow) * WSTR + ks * 32 + lhalf * 8;
      b[t].q[0] = *(const u32x4*)(br);
      b[t].q[1] = *(const u32x4*)(br + 16);
    }
#pragma unroll
    for (int t = 0; t < NTILES; ++t)
      acc[t] = __builtin_amdgcn_wmma_f32_16x16x32_bf16(
          false, a[ks].v, false, b[t].v, (short)0, acc[t], false, false);
  }

  // Epilogue: bias + activation + masked stores (fp32 and/or bf16 mirror).
#pragma unroll
  for (int t = 0; t < NTILES; ++t) {
    const int col = t * 16 + lrow;
    if (col >= out_cols) continue;
    const float bs = bias ? bias[col] : 0.f;
#pragma unroll
    for (int j = 0; j < 8; ++j) {
      const int row = r0 + j + lhalf * 8;
      if (row >= N) continue;
      float v = acc[t][j] + bs;
      if (act == 1) v = fmaxf(v, 0.f);
      else if (act == 2) v = 1.f / (1.f + __expf(-v));
      if (Yf) Yf[(size_t)row * ldo + col] = v;
      if (Yb) Yb[(size_t)row * DCH + col] = f2bf(v);
    }
  }
}

// ---------------------------------------------------------------------------
// kv[128,128] += K^T V (split-K over node chunks of 32, WMMA, atomic reduce).
// K/V chunks staged transposed (channel-major) in LDS so A and B lane loads
// are contiguous 16B/32B ds reads. All 8 B tiles preloaded per chunk.
// ---------------------------------------------------------------------------
__global__ __launch_bounds__(256) void k_kv(const unsigned short* __restrict__ Kb,
                                            const unsigned short* __restrict__ Vb,
                                            float* __restrict__ KV, int N) {
  constexpr int CSTR = 40;  // 80B rows, 16B aligned
  __shared__ __align__(16) unsigned short kt[DCH * CSTR];
  __shared__ __align__(16) unsigned short vt[DCH * CSTR];
  const int tid = threadIdx.x, wave = tid >> 5, lane = tid & 31;
  const int lhalf = lane >> 4, lrow = lane & 15;

  v8f acc[8];
#pragma unroll
  for (int t = 0; t < 8; ++t)
#pragma unroll
    for (int j = 0; j < 8; ++j) acc[t][j] = 0.f;

  const int nchunks = (N + 31) >> 5;
  for (int c = blockIdx.x; c < nchunks; c += gridDim.x) {
    __syncthreads();
    const int n0 = c << 5;
    for (int idx = tid; idx < 32 * DCH; idx += 256) {
      const int n = idx >> 7, d = idx & 127, row = n0 + n;
      unsigned short kk = 0, vv = 0;
      if (row < N) {
        kk = Kb[(size_t)row * DCH + d];
        vv = Vb[(size_t)row * DCH + d];
      }
      kt[d * CSTR + n] = kk;   // channel-major => A (=K^T) rows contiguous
      vt[d * CSTR + n] = vv;   // channel-major => B lane reads contiguous
    }
    __syncthreads();
    ABReg a;                   // A row d = wave*16 + lrow, 32 node values
    const unsigned short* ar = kt + (wave * 16 + lrow) * CSTR;
    a.q[0] = *(const u32x4*)(ar + lhalf * 8);
    a.q[1] = *(const u32x4*)(ar + 16 + lhalf * 8);
    ABReg b[8];
#pragma unroll
    for (int t = 0; t < 8; ++t) {
      const unsigned short* br = vt + (t * 16 + lrow) * CSTR + lhalf * 8;
      b[t].q[0] = *(const u32x4*)(br);
      b[t].q[1] = *(const u32x4*)(br + 16);
    }
#pragma unroll
    for (int t = 0; t < 8; ++t)
      acc[t] = __builtin_amdgcn_wmma_f32_16x16x32_bf16(
          false, a.v, false, b[t].v, (short)0, acc[t], false, false);
  }
#pragma unroll
  for (int t = 0; t < 8; ++t)
#pragma unroll
    for (int j = 0; j < 8; ++j)
      atomicAdd(&KV[(size_t)(wave * 16 + j + lhalf * 8) * DCH + t * 16 + lrow],
                acc[t][j]);
}

// ---------------------------------------------------------------------------
__global__ void k_fill(float* p, float v, size_t n) {
  size_t i = (size_t)blockIdx.x * blockDim.x + threadIdx.x;
  const size_t st = (size_t)gridDim.x * blockDim.x;
  for (; i < n; i += st) p[i] = v;
}

__global__ void k_deg(const long long* __restrict__ ei, float* __restrict__ deg,
                      int E) {
  int i = blockIdx.x * blockDim.x + threadIdx.x;
  const int st = gridDim.x * blockDim.x;
  for (; i < E; i += st) atomicAdd(&deg[(int)ei[(size_t)E + i]], 1.f);
}

__global__ void k_cvt(const float* __restrict__ x, unsigned short* __restrict__ xb,
                      size_t n, size_t npad) {
  size_t i = (size_t)blockIdx.x * blockDim.x + threadIdx.x;
  const size_t st = (size_t)gridDim.x * blockDim.x;
  for (; i < npad; i += st) xb[i] = (i < n) ? f2bf(x[i]) : (unsigned short)0;
}

// One wave per edge: lane moves float4 (512B/edge coalesced), f32 atomics.
// Prefetch the next grid-stride edge's source row to hide gather latency.
__global__ void k_gcn_edge(const long long* __restrict__ ei,
                           const float* __restrict__ deg,
                           const float* __restrict__ xw,
                           float* __restrict__ agg, int E) {
  const int wave = (blockIdx.x * blockDim.x + threadIdx.x) >> 5;
  const int lane = threadIdx.x & 31;
  const int nw = (gridDim.x * blockDim.x) >> 5;
  for (int e = wave; e < E; e += nw) {
    const int e2 = e + nw;
    if (e2 < E) {
      const int s2 = (int)ei[e2];
      __builtin_prefetch(xw + (size_t)s2 * DCH + lane * 4, 0, 0);
    }
    const int s = (int)ei[e], d = (int)ei[(size_t)E + e];
    const float coef = rsqrtf(deg[s]) * rsqrtf(deg[d]);
    const float4 v = *((const float4*)(xw + (size_t)s * DCH) + lane);
    float* dst = agg + (size_t)d * DCH + lane * 4;
    atomicAdd(dst + 0, v.x * coef);
    atomicAdd(dst + 1, v.y * coef);
    atomicAdd(dst + 2, v.z * coef);
    atomicAdd(dst + 3, v.w * coef);
  }
}

// Fused GCN combine + BN + relu + gated double-LayerNorm. One wave per row.
__global__ __launch_bounds__(256) void k_x1(
    const float* __restrict__ agg, const float* __restrict__ xw,
    const float* __restrict__ lin, const float* __restrict__ h,
    const float* __restrict__ deg, const float* __restrict__ gcn_b,
    const float* __restrict__ bn_mean, const float* __restrict__ bn_var,
    const float* __restrict__ bn_w, const float* __restrict__ bn_b,
    const float* __restrict__ ln0_g, const float* __restrict__ ln0_b,
    const float* __restrict__ lnf_g, const float* __restrict__ lnf_b,
    unsigned short* __restrict__ xgb, int N) {
  const int row = blockIdx.x * 8 + (threadIdx.x >> 5);
  if (row >= N) return;
  const int lane = threadIdx.x & 31;
  const float dinv = 1.f / deg[row];
  float x1[4], t[4];
#pragma unroll
  for (int j = 0; j < 4; ++j) {
    const int c = lane + 32 * j;
    const size_t o = (size_t)row * DCH + c;
    float a = agg[o] + xw[o] * dinv + gcn_b[c] + lin[o];
    a = (a - bn_mean[c]) * rsqrtf(bn_var[c] + EPS_C) * bn_w[c] + bn_b[c];
    x1[j] = fmaxf(a, 0.f);
    t[j] = h[o] * x1[j];
  }
  float mu = wred32(t[0] + t[1] + t[2] + t[3]) * (1.f / 128.f);
  float vs = 0.f;
#pragma unroll
  for (int j = 0; j < 4; ++j) { const float d0 = t[j] - mu; vs += d0 * d0; }
  const float rs = rsqrtf(wred32(vs) * (1.f / 128.f) + EPS_C);
  float xl[4];
#pragma unroll
  for (int j = 0; j < 4; ++j) {
    const int c = lane + 32 * j;
    xl[j] = ((t[j] - mu) * rs * ln0_g[c] + ln0_b[c]) * (1.f - BETA_C) +
            BETA_C * x1[j];
  }
  float mu2 = wred32(xl[0] + xl[1] + xl[2] + xl[3]) * (1.f / 128.f);
  float vs2 = 0.f;
#pragma unroll
  for (int j = 0; j < 4; ++j) { const float d0 = xl[j] - mu2; vs2 += d0 * d0; }
  const float rs2 = rsqrtf(wred32(vs2) * (1.f / 128.f) + EPS_C);
#pragma unroll
  for (int j = 0; j < 4; ++j) {
    const int c = lane + 32 * j;
    xgb[(size_t)row * DCH + c] = f2bf((xl[j] - mu2) * rs2 * lnf_g[c] + lnf_b[c]);
  }
}

__global__ void k_ksum(const unsigned short* __restrict__ Kb,
                       float* __restrict__ ks, int N) {
  const int c = threadIdx.x & 127, half = threadIdx.x >> 7;
  const int base = blockIdx.x * 64;
  const int lim = (base + 64 < N) ? base + 64 : N;
  float s = 0.f;
  for (int r = base + half; r < lim; r += 2) s += bf2f(Kb[(size_t)r * DCH + c]);
  atomicAdd(&ks[c], s);
}

// Attention epilogue: den = k.ksum, y = LN(num/den) * (hg + beta). Wave/row.
__global__ __launch_bounds__(256) void k_attn(
    const float* __restrict__ num, const unsigned short* __restrict__ Kb,
    const float* __restrict__ ks, const float* __restrict__ hg,
    const float* __restrict__ lng, const float* __restrict__ lnb,
    unsigned short* __restrict__ outb, int N) {
  const int row = blockIdx.x * 8 + (threadIdx.x >> 5);
  if (row >= N) return;
  const int lane = threadIdx.x & 31;
  float den = 0.f;
#pragma unroll
  for (int j = 0; j < 4; ++j) {
    const int c = lane + 32 * j;
    den += bf2f(Kb[(size_t)row * DCH + c]) * ks[c];
  }
  den = wred32(den);
  const float inv = 1.f / den;
  float y[4];
#pragma unroll
  for (int j = 0; j < 4; ++j)
    y[j] = num[(size_t)row * DCH + lane + 32 * j] * inv;
  const float mu = wred32(y[0] + y[1] + y[2] + y[3]) * (1.f / 128.f);
  float vs = 0.f;
#pragma unroll
  for (int j = 0; j < 4; ++j) { const float d0 = y[j] - mu; vs += d0 * d0; }
  const float rs = rsqrtf(wred32(vs) * (1.f / 128.f) + EPS_C);
#pragma unroll
  for (int j = 0; j < 4; ++j) {
    const int c = lane + 32 * j;
    const float z = (y[j] - mu) * rs * lng[c] + lnb[c];
    outb[(size_t)row * DCH + c] = f2bf(z * (hg[(size_t)row * DCH + c] + BETA_C));
  }
}

// ---------------------------------------------------------------------------
extern "C" void kernel_launch(void* const* d_in, const int* in_sizes, int n_in,
                              void* d_out, int out_size, void* d_ws,
                              size_t ws_size, hipStream_t stream) {
  const float* x          = (const float*)d_in[0];
  const long long* ei     = (const long long*)d_in[1];
  const float* h_lin_w    = (const float*)d_in[2];
  const float* h_lin_b    = (const float*)d_in[3];
  const float* gcn_w      = (const float*)d_in[4];
  const float* gcn_b      = (const float*)d_in[5];
  const float* lin_w      = (const float*)d_in[6];
  const float* lin_b      = (const float*)d_in[7];
  const float* bn_mean    = (const float*)d_in[8];
  const float* bn_var     = (const float*)d_in[9];
  const float* bn_w       = (const float*)d_in[10];
  const float* bn_b       = (const float*)d_in[11];
  const float* ln0_g      = (const float*)d_in[12];
  const float* ln0_b      = (const float*)d_in[13];
  const float* lnf_g      = (const float*)d_in[14];
  const float* lnf_b      = (const float*)d_in[15];
  const float* ga_h_w     = (const float*)d_in[16];
  const float* ga_h_b     = (const float*)d_in[17];
  const float* ga_k_w     = (const float*)d_in[18];
  const float* ga_k_b     = (const float*)d_in[19];
  const float* ga_v_w     = (const float*)d_in[20];
  const float* ga_v_b     = (const float*)d_in[21];
  const float* ga_ln_g    = (const float*)d_in[22];
  const float* ga_ln_b    = (const float*)d_in[23];
  const float* ga_out_w   = (const float*)d_in[24];
  const float* ga_out_b   = (const float*)d_in[25];
  const float* pred_w     = (const float*)d_in[26];
  const float* pred_b     = (const float*)d_in[27];

  const int N = in_sizes[0] / DCH;
  const int E = in_sizes[1] / 2;
  const int NP = ((N + 127) / 128) * 128;   // row padding for WMMA A loads
  const size_t npD = (size_t)NP * DCH;

  float* ws  = (float*)d_ws;
  float* f_a = ws;              // xw, later num
  float* f_b = f_a + npD;       // h,  later hg
  float* f_c = f_b + npD;       // lin_out
  float* f_d = f_c + npD;       // agg
  float* deg = f_d + npD;
  float* kv  = deg + NP;
  float* ks  = kv + DCH * DCH;
  unsigned short* b_x  = (unsigned short*)(ks + 128);  // x bf16, later attn-out
  unsigned short* b_xg = b_x + npD;
  unsigned short* b_k  = b_xg + npD;
  unsigned short* b_v  = b_k + npD;

  const int gRow = NP / 128;
  // local stage ------------------------------------------------------------
  k_fill<<<256, 256, 0, stream>>>(deg, 1.f, (size_t)N);      // self-loop
  k_deg<<<1024, 256, 0, stream>>>(ei, deg, E);
  k_cvt<<<2048, 256, 0, stream>>>(x, b_x, (size_t)N * DCH, npD);
  k_gemm<8><<<gRow, 256, 0, stream>>>(b_x, h_lin_w, h_lin_b, f_b, nullptr, N, DCH, DCH, 1);
  k_gemm<8><<<gRow, 256, 0, stream>>>(b_x, gcn_w, nullptr, f_a, nullptr, N, DCH, DCH, 0);
  k_gemm<8><<<gRow, 256, 0, stream>>>(b_x, lin_w, lin_b, f_c, nullptr, N, DCH, DCH, 0);
  k_fill<<<2048, 256, 0, stream>>>(f_d, 0.f, npD);
  k_gcn_edge<<<4096, 256, 0, stream>>>(ei, deg, f_a, f_d, E);
  k_x1<<<(N + 7) / 8, 256, 0, stream>>>(f_d, f_a, f_c, f_b, deg, gcn_b, bn_mean,
                                        bn_var, bn_w, bn_b, ln0_g, ln0_b, lnf_g,
                                        lnf_b, b_xg, N);
  // global linear-attention stage ------------------------------------------
  for (int i = 0; i < GLAYERS; ++i) {
    const size_t wo = (size_t)i * DCH * DCH, bo = (size_t)i * DCH;
    k_gemm<8><<<gRow, 256, 0, stream>>>(b_xg, ga_h_w + wo, ga_h_b + bo, f_b, nullptr, N, DCH, DCH, 0);
    k_gemm<8><<<gRow, 256, 0, stream>>>(b_xg, ga_k_w + wo, ga_k_b + bo, nullptr, b_k, N, DCH, DCH, 2);
    k_gemm<8><<<gRow, 256, 0, stream>>>(b_xg, ga_v_w + wo, ga_v_b + bo, nullptr, b_v, N, DCH, DCH, 0);
    k_fill<<<64, 256, 0, stream>>>(kv, 0.f, (size_t)(DCH * DCH + 128));
    k_ksum<<<(N + 63) / 64, 256, 0, stream>>>(b_k, ks, N);
    k_kv<<<208, 256, 0, stream>>>(b_k, b_v, kv, N);
    k_gemm<8><<<gRow, 256, 0, stream>>>(b_k, kv, nullptr, f_a, nullptr, N, DCH, DCH, 0);
    k_attn<<<(N + 7) / 8, 256, 0, stream>>>(f_a, b_k, ks, f_b, ga_ln_g + bo,
                                            ga_ln_b + bo, b_x, N);
    k_gemm<8><<<gRow, 256, 0, stream>>>(b_x, ga_out_w, ga_out_b, nullptr, b_xg, N, DCH, DCH, 1);
  }
  // prediction head: [128 -> 40], 3 column tiles, clamped stores -----------
  k_gemm<3><<<gRow, 256, 0, stream>>>(b_xg, pred_w, pred_b, (float*)d_out,
                                      nullptr, N, OUTC, OUTC, 0);
}
